// Attention_75402445849133
// MI455X (gfx1250) — compile-verified, matrix operand order checked by reading, not run on hardware
//
#include <hip/hip_runtime.h>

#define B_SZ 256
#define FIN  1024
#define HID  1024
#define NH   16
#define S_SZ 128
#define XTS  136   // x-tile LDS row stride (elems), 272B: 16B-aligned, bank-shifted
#define QS   72    // Q/K LDS row stride (elems), 144B: 16B-aligned, bank-shifted
#define ATS  136   // attn LDS row stride

typedef __attribute__((ext_vector_type(16))) __bf16 v16bf;
typedef __attribute__((ext_vector_type(8)))  float  v8f;

union Frag {
    v16bf v;
    uint4 q[2];
    unsigned int u[8];
};

// bf16 truncation (1 op)
__device__ __forceinline__ unsigned short f2bf(float f) {
    return (unsigned short)(__float_as_uint(f) >> 16);
}

// pack two fp32 -> packed bf16 pair via one v_perm_b32 (truncate)
__device__ __forceinline__ unsigned int pack_bf2(float lo, float hi) {
    return __builtin_amdgcn_perm(__float_as_uint(hi), __float_as_uint(lo), 0x07060302u);
}

// B-fragment (32x16 KxN) from fp32 row-major (k contiguous), bf16-converted via perm.
__device__ __forceinline__ Frag load_f32_frag(const float* __restrict__ p, int kh) {
    Frag f;
    const float4 a = *(const float4*)(p + kh);
    const float4 b = *(const float4*)(p + kh + 4);
    const float4 c = *(const float4*)(p + kh + 16);
    const float4 d = *(const float4*)(p + kh + 20);
    f.u[0] = pack_bf2(a.x, a.y); f.u[1] = pack_bf2(a.z, a.w);
    f.u[2] = pack_bf2(b.x, b.y); f.u[3] = pack_bf2(b.z, b.w);
    f.u[4] = pack_bf2(c.x, c.y); f.u[5] = pack_bf2(c.z, c.w);
    f.u[6] = pack_bf2(d.x, d.y); f.u[7] = pack_bf2(d.z, d.w);
    return f;
}

__device__ __forceinline__ v8f wmma_bf16(v16bf a, v16bf b, v8f c) {
    return __builtin_amdgcn_wmma_f32_16x16x32_bf16(
        false, a, false, b, (short)0, c, false, false);
}

// ---------- pre-pass: Wq/Wk fp32 -> bf16 into workspace ----------
__global__ __launch_bounds__(256)
void convert_w_kernel(const float4* __restrict__ Wq, const float4* __restrict__ Wk,
                      uint2* __restrict__ oq, uint2* __restrict__ ok)
{
    const int i = blockIdx.x * 256 + threadIdx.x;   // 262144 float4s per matrix
    const float4 a = Wq[i];
    const float4 b = Wk[i];
    uint2 pa, pb;
    pa.x = pack_bf2(a.x, a.y); pa.y = pack_bf2(a.z, a.w);
    pb.x = pack_bf2(b.x, b.y); pb.y = pack_bf2(b.z, b.w);
    oq[i] = pa;
    ok[i] = pb;
}

// ---------- fused attention ----------
__global__ __launch_bounds__(256)
void attn_fused_kernel(const float* __restrict__ x,
                       const unsigned short* __restrict__ Wqb,
                       const float* __restrict__ bq,
                       const unsigned short* __restrict__ Wkb,
                       const float* __restrict__ bk,
                       const float* __restrict__ Wo, const float* __restrict__ bo,
                       float* __restrict__ out)
{
    __shared__ __align__(16) unsigned short smem[18432]; // 36,864 B, phase-reused
    __shared__ float red[8 * 64];

    const int tid  = threadIdx.x;
    const int wave = tid >> 5;
    const int lane = tid & 31;
    const int col  = lane & 15;
    const int hlf  = lane >> 4;
    const int kh   = hlf ? 8 : 0;        // K-chunk select per ISA 16-bit A/B layout
    const int s0   = wave << 4;          // this wave's 16-row s strip

    const int b = blockIdx.x >> 4;
    const int h = blockIdx.x & 15;

    const float* __restrict__ xb = x + (size_t)b * FIN * S_SZ;

    unsigned short* xt   = smem;                 // phase A: 128 x XTS
    unsigned short* qlds = smem;                 // phase B: 128 x QS
    unsigned short* klds = smem + 128 * QS;      // phase B: 128 x QS
    unsigned short* attn = smem;                 // phase C: 128 x ATS

    // ---------------- Phase A: Q/K projection, tiled over i ----------------
    v8f qacc[4] = {};
    v8f kacc[4] = {};

    for (int it = 0; it < 8; ++it) {
        const int i0 = it << 7;
        __syncthreads();
        // Stage x[b, i0:i0+128, :] -> LDS bf16, transposed to [s][i], packed b32 stores
        #pragma unroll
        for (int r = 0; r < 8; ++r) {
            const int p  = (r << 8) + tid;       // 0..2047 (pairs of i-rows)
            const int ii = (p >> 5) << 1;        // even i in [0,128)
            const int s4 = (p & 31) << 2;        // s start
            const float4 a = *(const float4*)(xb + (size_t)(i0 + ii)     * S_SZ + s4);
            const float4 c = *(const float4*)(xb + (size_t)(i0 + ii + 1) * S_SZ + s4);
            *(unsigned int*)(xt + (s4 + 0) * XTS + ii) = pack_bf2(a.x, c.x);
            *(unsigned int*)(xt + (s4 + 1) * XTS + ii) = pack_bf2(a.y, c.y);
            *(unsigned int*)(xt + (s4 + 2) * XTS + ii) = pack_bf2(a.z, c.z);
            *(unsigned int*)(xt + (s4 + 3) * XTS + ii) = pack_bf2(a.w, c.w);
        }
        if (it < 7)
            __builtin_prefetch(xb + (size_t)(i0 + 128 + (tid >> 1)) * S_SZ + (tid & 1) * 64, 0, 0);
        __syncthreads();

        #pragma unroll
        for (int ks = 0; ks < 4; ++ks) {
            const int kk = ks << 5;
            Frag a;                               // A = x^T strip (16x32), shared by Q and K
            const unsigned short* arow = xt + (s0 + col) * XTS + kk;
            a.q[0] = *(const uint4*)(arow + kh);
            a.q[1] = *(const uint4*)(arow + kh + 16);
            #pragma unroll
            for (int ct = 0; ct < 4; ++ct) {
                const int wrow = (h << 6) + (ct << 4) + col;
                const unsigned short* wqp = Wqb + (size_t)wrow * FIN + i0 + kk;
                const unsigned short* wkp = Wkb + (size_t)wrow * FIN + i0 + kk;
                Frag bfq, bfk;
                bfq.q[0] = *(const uint4*)(wqp + kh);
                bfq.q[1] = *(const uint4*)(wqp + kh + 16);
                qacc[ct] = wmma_bf16(a.v, bfq.v, qacc[ct]);
                bfk.q[0] = *(const uint4*)(wkp + kh);
                bfk.q[1] = *(const uint4*)(wkp + kh + 16);
                kacc[ct] = wmma_bf16(a.v, bfk.v, kacc[ct]);
            }
        }
    }

    __syncthreads();  // all xt reads done before overwriting with Q/K
    #pragma unroll
    for (int ct = 0; ct < 4; ++ct) {
        const float bqv = bq[(h << 6) + (ct << 4) + col];
        const float bkv = bk[(h << 6) + (ct << 4) + col];
        #pragma unroll
        for (int v = 0; v < 8; ++v) {
            const int row = s0 + v + (hlf << 3);
            qlds[row * QS + (ct << 4) + col] = f2bf(qacc[ct][v] + bqv);
            klds[row * QS + (ct << 4) + col] = f2bf(kacc[ct][v] + bkv);
        }
    }
    __syncthreads();

    // ---------------- Phase B: scores = Q K^T ----------------
    v8f sacc[8] = {};
    #pragma unroll
    for (int ks = 0; ks < 2; ++ks) {
        const int kd = ks << 5;
        Frag a;
        const unsigned short* arow = qlds + (s0 + col) * QS + kd;
        a.q[0] = *(const uint4*)(arow + kh);
        a.q[1] = *(const uint4*)(arow + kh + 16);
        #pragma unroll
        for (int ct = 0; ct < 8; ++ct) {
            Frag bf;
            const unsigned short* brow = klds + ((ct << 4) + col) * QS + kd;
            bf.q[0] = *(const uint4*)(brow + kh);
            bf.q[1] = *(const uint4*)(brow + kh + 16);
            sacc[ct] = wmma_bf16(a.v, bf.v, sacc[ct]);
        }
    }

    // ---------------- Softmax over t ----------------
    float rmax[8], rinv[8];
    #pragma unroll
    for (int v = 0; v < 8; ++v) {
        float m = sacc[0][v];
        #pragma unroll
        for (int ct = 1; ct < 8; ++ct) m = fmaxf(m, sacc[ct][v]);
        #pragma unroll
        for (int d = 1; d < 16; d <<= 1) m = fmaxf(m, __shfl_xor(m, d, 32));
        rmax[v] = m;
    }
    #pragma unroll
    for (int v = 0; v < 8; ++v) {
        float s = 0.0f;
        #pragma unroll
        for (int ct = 0; ct < 8; ++ct) {
            const float e = __expf((sacc[ct][v] - rmax[v]) * 0.125f); // 1/TEMP folded in
            sacc[ct][v] = e;
            s += e;
        }
        #pragma unroll
        for (int d = 1; d < 16; d <<= 1) s += __shfl_xor(s, d, 32);
        rinv[v] = 1.0f / s;
    }

    __syncthreads();  // Q/K reads done before overwriting with attn
    #pragma unroll
    for (int ct = 0; ct < 8; ++ct)
        #pragma unroll
        for (int v = 0; v < 8; ++v) {
            const int row = s0 + v + (hlf << 3);
            attn[row * ATS + (ct << 4) + col] = f2bf(sacc[ct][v] * rinv[v]);
        }
    __syncthreads();

    // ---------------- Phase C: out = attn @ V (V = raw x head slice) ----------------
    v8f oacc[4] = {};
    #pragma unroll
    for (int ks = 0; ks < 4; ++ks) {
        const int kt = ks << 5;
        Frag a;
        const unsigned short* arow = attn + (s0 + col) * ATS + kt;
        a.q[0] = *(const uint4*)(arow + kh);
        a.q[1] = *(const uint4*)(arow + kh + 16);
        #pragma unroll
        for (int ct = 0; ct < 4; ++ct) {
            const float* xv = xb + (size_t)((h << 6) + (ct << 4) + col) * S_SZ + kt;
            Frag bf = load_f32_frag(xv, kh);
            oacc[ct] = wmma_bf16(a.v, bf.v, oacc[ct]);
        }
    }

    // ---------------- Final: score[b, h*64+d] = sum_s out[s,d] * Wo[s] + bo ----------------
    float wo[8];
    #pragma unroll
    for (int v = 0; v < 8; ++v) wo[v] = Wo[s0 + v + (hlf << 3)];
    #pragma unroll
    for (int ct = 0; ct < 4; ++ct) {
        float p = 0.0f;
        #pragma unroll
        for (int v = 0; v < 8; ++v) p += oacc[ct][v] * wo[v];
        p += __shfl_xor(p, 16, 32);               // merge the two M-halves (same d)
        if (hlf == 0) red[wave * 64 + (ct << 4) + col] = p;
    }
    __syncthreads();
    if (tid < 64) {
        float s = 0.0f;
        #pragma unroll
        for (int w = 0; w < 8; ++w) s += red[w * 64 + tid];
        out[(size_t)b * HID + (h << 6) + tid] = s + bo[0];
    }
}

extern "C" void kernel_launch(void* const* d_in, const int* in_sizes, int n_in,
                              void* d_out, int out_size, void* d_ws, size_t ws_size,
                              hipStream_t stream) {
    const float* x  = (const float*)d_in[0];
    const float* Wq = (const float*)d_in[1];
    const float* bq = (const float*)d_in[2];
    const float* Wk = (const float*)d_in[3];
    const float* bk = (const float*)d_in[4];
    const float* Wo = (const float*)d_in[5];
    const float* bo = (const float*)d_in[6];
    float* out = (float*)d_out;

    // workspace: Wq_bf16 (2 MB) | Wk_bf16 (2 MB)
    unsigned short* wqb = (unsigned short*)d_ws;
    unsigned short* wkb = wqb + (size_t)HID * FIN;

    convert_w_kernel<<<dim3((HID * FIN / 4) / 256), dim3(256), 0, stream>>>(
        (const float4*)Wq, (const float4*)Wk, (uint2*)wqb, (uint2*)wkb);

    attn_fused_kernel<<<dim3(B_SZ * NH), dim3(256), 0, stream>>>(
        x, wqb, bq, wkb, bk, Wo, bo, out);

    (void)in_sizes; (void)n_in; (void)out_size; (void)ws_size;
}
